// GraspFast_82617990906312
// MI455X (gfx1250) — matrix operand order repaired and checked by reading, not compiled, not run on hardware
//
#include <hip/hip_runtime.h>
#include <hip/hip_bf16.h>

typedef __attribute__((ext_vector_type(16))) _Float16 v16h;
typedef __attribute__((ext_vector_type(8)))  float    v8f;

#define BATCH 4
#define NPTS  20000
#define CFEAT 512
#define MPT   1024
#define NS    16
#define F1    515       // 3 + 512
#define K1PAD 544       // 17 * 32
#define KT1   17
#define KT2   8
#define KTF   32
#define BIGF  1.0e10f

// ---------------- workspace layout (bytes) ----------------
static constexpr size_t OFF_IDXS  = 0;                        // [4,1024] i32
static constexpr size_t OFF_XYZG  = 16384;                    // [4,1024,3] f32
static constexpr size_t OFF_FEATG = 65536;                    // [4,1024,512] f16
static constexpr size_t OFF_GIDX  = 4259840;                  // [4,4,1024,16] i32
static constexpr size_t OFF_GXYZ  = 5308416;                  // [4,4,1024,16,3] f16
static constexpr size_t OFF_W1P   = 6881280;                  // [4,16,17,512] f16 (A-fragments, K-permuted)
static constexpr size_t OFF_W2P   = 7995392;                  // [4,16,8,512]  f16
static constexpr size_t OFF_FWP   = 8519680;                  // [16,32,512]   f16
static constexpr size_t OFF_CAT   = 9043968;                  // [4,1024,1024] f16

__device__ inline v8f wmma16(v16h a, v16h b, v8f c) {
  return __builtin_amdgcn_wmma_f32_16x16x32_f16(false, a, false, b, (short)0, c, false, false);
}

// ---------------- Tensor Data Mover helper (gfx1250) ----------------
#if __has_builtin(__builtin_amdgcn_tensor_load_to_lds)
#define HAVE_TDM 1
typedef __attribute__((ext_vector_type(4))) unsigned tdm_v4u;
typedef __attribute__((ext_vector_type(8))) int      tdm_v8i;
typedef __attribute__((ext_vector_type(4))) int      tdm_v4i;
__device__ inline void tdm_load(tdm_v4u g0, tdm_v8i g1, tdm_v4i g2, tdm_v4i g3) {
#if defined(__clang_major__) && __clang_major__ >= 23
  __builtin_amdgcn_tensor_load_to_lds(g0, g1, g2, g3, (tdm_v8i){}, 0);
#else
  __builtin_amdgcn_tensor_load_to_lds(g0, g1, g2, g3, 0);
#endif
}
#endif

// ---------------------------------------------------------------------------
// Pack an [O x K] f32 row-major matrix into WMMA 16x16x32 f16 A-fragments.
// Fragment (mt, kt): 32 lanes x 16 halves, lane l element e:
//   row = l & 15 ;  k = (l>>4)*16 + e
// permute!=0: layer-1 K reorder (feat channels first, xyz last):
//   ksrc = (k < 512) ? k + 3 : k - 512     (k in [0, 515))
// ---------------------------------------------------------------------------
__global__ __launch_bounds__(256) void pack_a_kernel(const float* __restrict__ src,
                                                     _Float16* __restrict__ dst,
                                                     int O, int K, int Kpad, int permute) {
  int mtc = O >> 4, ktc = Kpad >> 5;
  int bid = blockIdx.x;
  int mat = bid / (mtc * ktc);
  int rem = bid - mat * (mtc * ktc);
  int mt = rem / ktc, kt = rem - mt * ktc;
  size_t base = ((size_t)bid) * 512;
  const float* s = src + (size_t)mat * O * K;
  for (int i = threadIdx.x; i < 512; i += 256) {
    int lane = i >> 4, e = i & 15;
    int hi = lane >> 4;
    int row = mt * 16 + (lane & 15);
    int kc  = hi * 16 + e;
    int k   = kt * 32 + kc;
    float v = 0.0f;
    if (k < K) {
      int ksrc = permute ? ((k < 512) ? (k + 3) : (k - 512)) : k;
      v = s[(size_t)row * K + ksrc];
    }
    dst[base + i] = (_Float16)v;
  }
}

// ---------------------------------------------------------------------------
// Masked furthest point sampling. One 1024-thread workgroup per batch.
// ---------------------------------------------------------------------------
__global__ __launch_bounds__(1024) void fps_kernel(const float* __restrict__ xyz,
                                                   const float* __restrict__ obj,
                                                   const float* __restrict__ grasp,
                                                   int* __restrict__ idxs) {
  const int b = blockIdx.x, tid = threadIdx.x;
  const int lane = tid & 31, wid = tid >> 5;
  __shared__ float wd[32];
  __shared__ int   wn[32];
  __shared__ int   sWin;

  const float* xb = xyz + (size_t)b * NPTS * 3;
  const float* o0 = obj + (size_t)b * 2 * NPTS;
  const float* o1 = o0 + NPTS;
  const float* gb = grasp + (size_t)b * NPTS;

  float d[20];
  int firstn = 0x7fffffff;
  #pragma unroll
  for (int s = 0; s < 20; ++s) {
    int n = s * 1024 + tid;
    float dv = -BIGF;
    if (n < NPTS) {
      bool m = (o1[n] > o0[n]) && (gb[n] > 0.1f);
      dv = m ? BIGF : -BIGF;
      if (m && n < firstn) firstn = n;
    }
    d[s] = dv;
  }
  #pragma unroll
  for (int off = 16; off > 0; off >>= 1) firstn = min(firstn, __shfl_xor(firstn, off, 32));
  if (lane == 0) wn[wid] = firstn;
  __syncthreads();
  if (wid == 0) {
    int v = wn[lane];
    #pragma unroll
    for (int off = 16; off > 0; off >>= 1) v = min(v, __shfl_xor(v, off, 32));
    if (lane == 0) {
      int f = (v == 0x7fffffff) ? 0 : v;
      sWin = f;
      idxs[b * MPT] = f;
    }
  }
  __syncthreads();
  int prev = sWin;

  for (int i = 1; i < MPT; ++i) {
    __syncthreads();                       // everyone consumed sWin
    float lx = xb[prev * 3 + 0], ly = xb[prev * 3 + 1], lz = xb[prev * 3 + 2];
    float bd = -3.0e38f;
    int bn = 0x7fffffff;
    #pragma unroll
    for (int s = 0; s < 20; ++s) {
      int n = s * 1024 + tid;
      if (n < NPTS) {
        float dv = d[s];
        if (dv > -BIGF) {
          float dx = xb[n * 3 + 0] - lx;
          float dy = xb[n * 3 + 1] - ly;
          float dz = xb[n * 3 + 2] - lz;
          dv = fminf(dv, dx * dx + dy * dy + dz * dz);
          d[s] = dv;
        }
        if (dv > bd || (dv == bd && n < bn)) { bd = dv; bn = n; }
      }
    }
    #pragma unroll
    for (int off = 16; off > 0; off >>= 1) {
      float od = __shfl_xor(bd, off, 32);
      int   on = __shfl_xor(bn, off, 32);
      if (od > bd || (od == bd && on < bn)) { bd = od; bn = on; }
    }
    if (lane == 0) { wd[wid] = bd; wn[wid] = bn; }
    __syncthreads();
    if (wid == 0) {
      float v = wd[lane];
      int  nn = wn[lane];
      #pragma unroll
      for (int off = 16; off > 0; off >>= 1) {
        float od = __shfl_xor(v, off, 32);
        int   on = __shfl_xor(nn, off, 32);
        if (od > v || (od == v && on < nn)) { v = od; nn = on; }
      }
      if (lane == 0) { sWin = nn; idxs[b * MPT + i] = nn; }
    }
    __syncthreads();
    prev = sWin;
  }
}

// ---------------------------------------------------------------------------
// Gather sampled xyz (f32) and features (f16, [b][m][c]).
// ---------------------------------------------------------------------------
__global__ __launch_bounds__(256) void gather_kernel(const float* __restrict__ xyz,
                                                     const float* __restrict__ feat,
                                                     const int* __restrict__ idxs,
                                                     float* __restrict__ xyz_g,
                                                     _Float16* __restrict__ feat_g) {
  int bm = blockIdx.x, b = bm >> 10, tid = threadIdx.x;
  int row = idxs[bm];
  if (tid < 3) xyz_g[bm * 3 + tid] = xyz[((size_t)b * NPTS + row) * 3 + tid];
  const float* fb = feat + (size_t)b * CFEAT * NPTS + row;
  _Float16* dst = feat_g + (size_t)bm * CFEAT;
  for (int c = tid; c < CFEAT; c += 256) dst[c] = (_Float16)fb[(size_t)c * NPTS];
}

// ---------------------------------------------------------------------------
// Cylinder query: first-16-in-index-order neighbor selection per (r,b,m).
// ---------------------------------------------------------------------------
__global__ __launch_bounds__(128) void query_kernel(const float* __restrict__ xyz_g,
                                                    const float* __restrict__ rot,
                                                    int* __restrict__ gidx,
                                                    _Float16* __restrict__ gxyz) {
  __shared__ float sx[MPT], sy[MPT], sz[MPT];
  __shared__ int scnt[128];
  __shared__ int sel[16];

  int bid = blockIdx.x;
  int r = bid >> 12, bm = bid & 4095, b = bm >> 10, m = bm & 1023;
  int tid = threadIdx.x;
  float rad = (r == 0) ? 0.025f : (r == 1) ? 0.05f : (r == 2) ? 0.075f : 0.1f;

  const float* xb = xyz_g + (size_t)b * MPT * 3;
  for (int n = tid; n < MPT; n += 128) {
    sx[n] = xb[n * 3 + 0]; sy[n] = xb[n * 3 + 1]; sz[n] = xb[n * 3 + 2];
  }
  __syncthreads();

  const float* R = rot + ((size_t)b * MPT + m) * 9;
  float r00 = R[0], r01 = R[1], r02 = R[2];
  float r10 = R[3], r11 = R[4], r12 = R[5];
  float r20 = R[6], r21 = R[7], r22 = R[8];
  float cx = sx[m], cy = sy[m], cz = sz[m];
  float rad2 = rad * rad;

  unsigned flags = 0;
  int cnt = 0, n0 = tid * 8;
  #pragma unroll
  for (int j = 0; j < 8; ++j) {
    int n = n0 + j;
    float dx = sx[n] - cx, dy = sy[n] - cy, dz = sz[n] - cz;
    float l0 = dx * r00 + dy * r10 + dz * r20;
    float l1 = dx * r01 + dy * r11 + dz * r21;
    float l2 = dx * r02 + dy * r12 + dz * r22;
    float rr = l1 * l1 + l2 * l2;
    if ((rr < rad2) && (l0 > -0.02f) && (l0 < 0.04f)) { flags |= (1u << j); ++cnt; }
  }
  scnt[tid] = cnt;
  __syncthreads();
  for (int s2 = 1; s2 < 128; s2 <<= 1) {           // inclusive scan
    int v = scnt[tid];
    int a = (tid >= s2) ? scnt[tid - s2] : 0;
    __syncthreads();
    scnt[tid] = v + a;
    __syncthreads();
  }
  int total = scnt[127];
  int pref = scnt[tid] - cnt;
  #pragma unroll
  for (int j = 0; j < 8; ++j) {
    if (flags & (1u << j)) {
      if (pref < 16) sel[pref] = n0 + j;
      ++pref;
    }
  }
  __syncthreads();
  if (tid < 16) {
    int p = (total == 0) ? 0 : ((tid < total) ? sel[tid] : sel[0]);
    float dx = sx[p] - cx, dy = sy[p] - cy, dz = sz[p] - cz;
    float l0 = dx * r00 + dy * r10 + dz * r20;
    float l1 = dx * r01 + dy * r11 + dz * r21;
    float l2 = dx * r02 + dy * r12 + dz * r22;
    size_t base = (((size_t)r * BATCH + b) * MPT + m) * NS + tid;
    gidx[base] = p;
    _Float16* g = gxyz + base * 3;
    float inv = 1.0f / rad;
    g[0] = (_Float16)(l0 * inv);
    g[1] = (_Float16)(l1 * inv);
    g[2] = (_Float16)(l2 * inv);
  }
}

// ---------------------------------------------------------------------------
// Grouped two-layer MLP + maxpool via WMMA. One block per (radius, b, m).
// 8 waves; wave w owns output M-tiles {2w, 2w+1}; N-tile = the 16 samples.
// Activation K layout (matches permuted w1 pack): k 0..511 = feat channels,
// k 512..514 = local xyz / radius, k 515..543 = zero pad.
// The 16 scattered feature rows are pulled into LDS by the Tensor Data Mover
// in gather mode (16 x 16-bit row indices, 1024B rows, LDS pad 16 dwords to
// land on the 544-half row stride).
// ---------------------------------------------------------------------------
__global__ __launch_bounds__(256) void mlp_kernel(const _Float16* __restrict__ feat_g,
                                                  const int* __restrict__ gidx,
                                                  const _Float16* __restrict__ gxyz,
                                                  const _Float16* __restrict__ w1p,
                                                  const _Float16* __restrict__ w2p,
                                                  const float* __restrict__ b1,
                                                  const float* __restrict__ b2,
                                                  _Float16* __restrict__ cat) {
  __shared__ __align__(32) _Float16 Bg[16][K1PAD];   // activations [n][k]
  __shared__ __align__(32) _Float16 H1[16][256];     // hidden      [n][o]
  __shared__ int sIdx[16];

  int bid = blockIdx.x;
  int r = bid >> 12, bm = bid & 4095, b = bm >> 10, m = bm & 1023;
  int tid = threadIdx.x;
  size_t gbase = (((size_t)r * BATCH + b) * MPT + m) * NS;

  if (tid < 16) sIdx[tid] = gidx[gbase + tid];
  __syncthreads();

  const _Float16* fg = feat_g + (size_t)b * MPT * CFEAT;

#ifdef HAVE_TDM
  // tail region: xyz (k=512..514) and zero pad (k=515..543)
  for (int pos = tid; pos < 16 * 32; pos += 256) {
    int n = pos >> 5, k = 512 + (pos & 31);
    Bg[n][k] = (k < F1) ? gxyz[(gbase + n) * 3 + (k - 512)] : (_Float16)0.0f;
  }
  if (tid < 32) {  // wave 0 drives the TDM gather of the 16 feature rows
    unsigned lds = (unsigned)(size_t)(void*)&Bg[0][0];
    unsigned long long ga = (unsigned long long)(size_t)(const void*)fg;
    tdm_v4u g0 = { 0x80000001u,                    // count=1 | gather_mode, 16-bit idx
                   lds,
                   (unsigned)ga,
                   (unsigned)((ga >> 32) & 0x01FFFFFFu) | 0x80000000u };  // type=2
    tdm_v8i g1 = { (int)0x1FD10000u,               // data_size=2B, pad_en, intv=256dw, amt=16dw
                   (int)(512u << 16),              // tensor_dim0 = 512 elems
                   (int)(1024u << 16),             // tensor_dim1 = 1024 rows
                   (int)(512u << 16),              // tile_dim0 = 512 elems (1024B row)
                   16,                             // tile_dim1 = 16 valid indices
                   512, 0, 0 };                    // tensor_dim0_stride = 512
    tdm_v4i g2 = { (sIdx[1]  << 16) | sIdx[0],  (sIdx[3]  << 16) | sIdx[2],
                   (sIdx[5]  << 16) | sIdx[4],  (sIdx[7]  << 16) | sIdx[6] };
    tdm_v4i g3 = { (sIdx[9]  << 16) | sIdx[8],  (sIdx[11] << 16) | sIdx[10],
                   (sIdx[13] << 16) | sIdx[12], (sIdx[15] << 16) | sIdx[14] };
    tdm_load(g0, g1, g2, g3);
    __builtin_amdgcn_s_wait_tensorcnt(0);
  }
#else
  for (int pos = tid; pos < 16 * K1PAD; pos += 256) {
    int n = pos / K1PAD, k = pos - n * K1PAD;
    _Float16 v;
    if (k < 512)      v = fg[(size_t)sIdx[n] * CFEAT + k];
    else if (k < F1)  v = gxyz[(gbase + n) * 3 + (k - 512)];
    else              v = (_Float16)0.0f;
    Bg[n][k] = v;
  }
#endif
  __syncthreads();

  int lane = tid & 31, w = tid >> 5;
  int nb = lane & 15, koffB = (lane >> 4) * 16;
  int mt0 = 2 * w, mt1 = 2 * w + 1;

  // ---- layer 1: [256 x 544] x [544 x 16] ----
  const _Float16* A1 = w1p + (size_t)r * 16 * KT1 * 512;
  v8f acc0 = {}, acc1 = {};
  #pragma unroll
  for (int kt = 0; kt < KT1; ++kt) {
    v16h bf = *(const v16h*)&Bg[nb][kt * 32 + koffB];
    const _Float16* p0 = A1 + (((size_t)mt0 * KT1 + kt) * 32 + lane) * 16;
    const _Float16* p1 = A1 + (((size_t)mt1 * KT1 + kt) * 32 + lane) * 16;
    __builtin_prefetch(p0 + 512, 0, 3);
    v16h a0 = *(const v16h*)p0;
    v16h a1 = *(const v16h*)p1;
    acc0 = wmma16(a0, bf, acc0);
    acc1 = wmma16(a1, bf, acc1);
  }
  {
    int colh = lane & 15, rbase = 8 * (lane >> 4);
    const float* bb = b1 + r * 256;
    #pragma unroll
    for (int e = 0; e < 8; ++e) {
      int o0 = mt0 * 16 + rbase + e;
      int o1 = mt1 * 16 + rbase + e;
      H1[colh][o0] = (_Float16)fmaxf(acc0[e] + bb[o0], 0.0f);
      H1[colh][o1] = (_Float16)fmaxf(acc1[e] + bb[o1], 0.0f);
    }
  }
  __syncthreads();

  // ---- layer 2: [256 x 256] x [256 x 16] ----
  const _Float16* A2 = w2p + (size_t)r * 16 * KT2 * 512;
  v8f c0 = {}, c1 = {};
  #pragma unroll
  for (int kt = 0; kt < KT2; ++kt) {
    v16h bf = *(const v16h*)&H1[nb][kt * 32 + koffB];
    v16h a0 = *(const v16h*)(A2 + (((size_t)mt0 * KT2 + kt) * 32 + lane) * 16);
    v16h a1 = *(const v16h*)(A2 + (((size_t)mt1 * KT2 + kt) * 32 + lane) * 16);
    c0 = wmma16(a0, bf, c0);
    c1 = wmma16(a1, bf, c1);
  }

  // ---- bias + relu + maxpool over the 16 samples (16-lane column groups) ----
  const float* bb2 = b2 + r * 256;
  int rbase = 8 * (lane >> 4);
  float v0[8], v1[8];
  #pragma unroll
  for (int e = 0; e < 8; ++e) {
    v0[e] = fmaxf(c0[e] + bb2[mt0 * 16 + rbase + e], 0.0f);
    v1[e] = fmaxf(c1[e] + bb2[mt1 * 16 + rbase + e], 0.0f);
  }
  #pragma unroll
  for (int e = 0; e < 8; ++e) {
    #pragma unroll
    for (int off = 1; off < 16; off <<= 1) {
      v0[e] = fmaxf(v0[e], __shfl_xor(v0[e], off, 32));
      v1[e] = fmaxf(v1[e], __shfl_xor(v1[e], off, 32));
    }
  }
  if ((lane & 15) == 0) {
    _Float16* crow = cat + ((size_t)b * MPT + m) * 1024 + r * 256;
    #pragma unroll
    for (int e = 0; e < 8; ++e) {
      crow[mt0 * 16 + rbase + e] = (_Float16)v0[e];
      crow[mt1 * 16 + rbase + e] = (_Float16)v1[e];
    }
  }
}

// ---------------------------------------------------------------------------
// Fuse GEMM: out[b,o,m] = fuse_w[o,:] . cat[b,m,:] + fuse_b[o] via WMMA.
// One block per (b, 16-point tile). 8 waves x 2 M-tiles x 32 K-steps.
// The 16x2048B activation block is staged into LDS with a plain 2-D TDM tile.
// ---------------------------------------------------------------------------
__global__ __launch_bounds__(256) void fuse_kernel(const _Float16* __restrict__ cat,
                                                   const _Float16* __restrict__ fwp,
                                                   const float* __restrict__ fb,
                                                   float* __restrict__ out) {
  __shared__ __align__(32) _Float16 Bc[16][1024];
  int bid = blockIdx.x, b = bid >> 6, nt = bid & 63;
  int tid = threadIdx.x;

  const _Float16* cb = cat + ((size_t)b * MPT + nt * 16) * 1024;
#ifdef HAVE_TDM
  if (tid < 32) {  // wave 0 drives the TDM block load
    unsigned lds = (unsigned)(size_t)(void*)&Bc[0][0];
    unsigned long long ga = (unsigned long long)(size_t)(const void*)cb;
    tdm_v4u g0 = { 1u,                              // count=1, normal mode
                   lds,
                   (unsigned)ga,
                   (unsigned)((ga >> 32) & 0x01FFFFFFu) | 0x80000000u };  // type=2
    tdm_v8i g1 = { (int)0x00010000u,                // data_size = 2B
                   (int)(1024u << 16),              // tensor_dim0 = 1024 elems
                   (int)(16u << 16),                // tensor_dim1 = 16 rows
                   (int)(1024u << 16),              // tile_dim0 = 1024 elems
                   16,                              // tile_dim1 = 16 rows
                   1024, 0, 0 };                    // tensor_dim0_stride = 1024
    tdm_v4i g2 = { 0, 0, 0, 0 };
    tdm_v4i g3 = { 0, 0, 0, 0 };
    tdm_load(g0, g1, g2, g3);
    __builtin_amdgcn_s_wait_tensorcnt(0);
  }
#else
  for (int chunk = tid; chunk < (16 * 1024) / 8; chunk += 256) {
    int off = chunk * 8;
    int n = off >> 10, k = off & 1023;
    *(uint4*)&Bc[n][k] = *(const uint4*)&cb[(size_t)n * 1024 + k];
  }
#endif
  __syncthreads();

  int lane = tid & 31, w = tid >> 5;
  int nb = lane & 15, koffB = (lane >> 4) * 16;
  int mt0 = 2 * w, mt1 = 2 * w + 1;
  v8f c0 = {}, c1 = {};
  #pragma unroll
  for (int kt = 0; kt < KTF; ++kt) {
    v16h bf = *(const v16h*)&Bc[nb][kt * 32 + koffB];
    const _Float16* p0 = fwp + (((size_t)mt0 * KTF + kt) * 32 + lane) * 16;
    const _Float16* p1 = fwp + (((size_t)mt1 * KTF + kt) * 32 + lane) * 16;
    __builtin_prefetch(p0 + 512, 0, 3);
    v16h a0 = *(const v16h*)p0;
    v16h a1 = *(const v16h*)p1;
    c0 = wmma16(a0, bf, c0);
    c1 = wmma16(a1, bf, c1);
  }
  int rbase = 8 * (lane >> 4), col = nt * 16 + (lane & 15);
  float* ob = out + (size_t)b * 256 * MPT;
  #pragma unroll
  for (int e = 0; e < 8; ++e) {
    int o0 = mt0 * 16 + rbase + e;
    int o1 = mt1 * 16 + rbase + e;
    ob[(size_t)o0 * MPT + col] = c0[e] + fb[o0];
    ob[(size_t)o1 * MPT + col] = c1[e] + fb[o1];
  }
}

// ---------------------------------------------------------------------------
extern "C" void kernel_launch(void* const* d_in, const int* in_sizes, int n_in,
                              void* d_out, int out_size, void* d_ws, size_t ws_size,
                              hipStream_t stream) {
  (void)in_sizes; (void)n_in; (void)out_size; (void)ws_size;
  const float* seed_xyz  = (const float*)d_in[0];
  const float* seed_feat = (const float*)d_in[1];
  const float* obj       = (const float*)d_in[2];
  const float* grasp     = (const float*)d_in[3];
  const float* rot       = (const float*)d_in[4];
  const float* w1        = (const float*)d_in[5];
  const float* b1        = (const float*)d_in[6];
  const float* w2        = (const float*)d_in[7];
  const float* b2        = (const float*)d_in[8];
  const float* fw        = (const float*)d_in[9];
  const float* fb        = (const float*)d_in[10];
  float* out = (float*)d_out;

  char* ws = (char*)d_ws;
  int*      idxs   = (int*)(ws + OFF_IDXS);
  float*    xyz_g  = (float*)(ws + OFF_XYZG);
  _Float16* feat_g = (_Float16*)(ws + OFF_FEATG);
  int*      gidx   = (int*)(ws + OFF_GIDX);
  _Float16* gxyz   = (_Float16*)(ws + OFF_GXYZ);
  _Float16* w1p    = (_Float16*)(ws + OFF_W1P);
  _Float16* w2p    = (_Float16*)(ws + OFF_W2P);
  _Float16* fwp    = (_Float16*)(ws + OFF_FWP);
  _Float16* cat    = (_Float16*)(ws + OFF_CAT);

  // weight packing (independent of data path, stream-ordered anyway)
  pack_a_kernel<<<4 * 16 * KT1, 256, 0, stream>>>(w1, w1p, 256, F1, K1PAD, 1);
  pack_a_kernel<<<4 * 16 * KT2, 256, 0, stream>>>(w2, w2p, 256, 256, 256, 0);
  pack_a_kernel<<<16 * KTF,     256, 0, stream>>>(fw, fwp, 256, 1024, 1024, 0);

  fps_kernel   <<<BATCH,        1024, 0, stream>>>(seed_xyz, obj, grasp, idxs);
  gather_kernel<<<BATCH * MPT,  256,  0, stream>>>(seed_xyz, seed_feat, idxs, xyz_g, feat_g);
  query_kernel <<<4 * BATCH * MPT, 128, 0, stream>>>(xyz_g, rot, gidx, gxyz);
  mlp_kernel   <<<4 * BATCH * MPT, 256, 0, stream>>>(feat_g, gidx, gxyz, w1p, w2p, b1, b2, cat);
  fuse_kernel  <<<BATCH * 64,   256,  0, stream>>>(cat, fwp, fb, out);
}
